// WindowAttention3D_FAST_66099546685505
// MI455X (gfx1250) — matrix operand order repaired and checked by reading, not compiled, not run on hardware
//
#include <hip/hip_runtime.h>

typedef __attribute__((ext_vector_type(16))) _Float16 v16h;
typedef __attribute__((ext_vector_type(4)))  _Float16 v4h;
typedef __attribute__((ext_vector_type(8)))  float    v8f;

#define NTOK 256   // tokens per window (4*8*8)
#define CDIM 128   // embedding dim
#define HD   32    // head dim
#define NH   4     // heads
#define NWIN 64    // shift-mask windows per volume

// ---- fragment helpers ------------------------------------------------------
// A-fragment (16x32 f16): per lane two contiguous 8-half chunks.
__device__ __forceinline__ v16h lds_afrag(const _Float16* p0, const _Float16* p1) {
  v16h r;
#pragma unroll
  for (int i = 0; i < 8; ++i) { r[i] = p0[i]; r[8 + i] = p1[i]; }
  return r;
}
// B-fragment (32x16 f16): per lane 16 contiguous halfs (2x ds_load_b128).
__device__ __forceinline__ v16h lds_bfrag(const _Float16* p) {
  v16h r;
#pragma unroll
  for (int i = 0; i < 16; ++i) r[i] = p[i];
  return r;
}
// A-fragment from global f32 with on-the-fly f16 conversion.
__device__ __forceinline__ v16h afrag_f32(const float* p0, const float* p1) {
  v16h r;
#pragma unroll
  for (int i = 0; i < 8; ++i) { r[i] = (_Float16)p0[i]; r[8 + i] = (_Float16)p1[i]; }
  return r;
}
__device__ __forceinline__ v8f wmma(v16h a, v16h b, v8f c) {
  return __builtin_amdgcn_wmma_f32_16x16x32_f16(false, a, false, b, (short)0, c, false, false);
}

// CDNA5 async DMA: 16 bytes global -> LDS per lane, tracked with ASYNCcnt.
__device__ __forceinline__ void async_g2l_b128(unsigned lds_addr, const void* gaddr) {
  asm volatile("global_load_async_to_lds_b128 %0, %1, off"
               :: "v"(lds_addr), "v"((unsigned long long)gaddr) : "memory");
}
__device__ __forceinline__ void wait_asynccnt0() {
#if __has_builtin(__builtin_amdgcn_s_wait_asynccnt)
  __builtin_amdgcn_s_wait_asynccnt(0);
#else
  asm volatile("s_wait_asynccnt 0x0" ::: "memory");
#endif
}
// Generic (shared) pointer -> 32-bit LDS byte address for the async DMA VDST.
__device__ __forceinline__ unsigned lds_byte_addr(const void* p) {
  return (unsigned)(unsigned long long)
      (__attribute__((address_space(3))) const void*)p;
}

// Fused 3D window attention: one workgroup per (window b, head h).
// 512 threads = 16 wave32's; wave w owns query rows [16w, 16w+16).
__global__ __launch_bounds__(512, 1)
void win_attn3d_kernel(const float* __restrict__ x,      // (512, 256, 128)
                       const float* __restrict__ mask,   // (64, 256, 256)
                       const float* __restrict__ qkvw,   // (384, 128)
                       const float* __restrict__ qb,     // (128,)
                       const float* __restrict__ vb,     // (128,)
                       const float* __restrict__ table,  // (1575, 4)
                       float*       __restrict__ out)    // (512, 256, 128)
{
  // f32 weight staging (phase 0 only) aliases Q/K/V (written from phase 1 on).
  __shared__ __align__(16) union {
    float stage[96 * CDIM];                     // 48 KB
    struct {
      _Float16 q[NTOK * HD];                    // Q row-major, pre-scaled
      _Float16 k[NTOK * HD];                    // K row-major
      _Float16 v[HD * NTOK];                    // V transposed [d][m]
    } qkv;                                      // 48 KB
  } uS;
  __shared__ __align__(16) _Float16 sW16[96 * CDIM];   // f16 weight cache (24 KB)
  __shared__ __align__(16) _Float16 sP[16][16 * 32];   // per-wave P staging (16 KB)

  const int b   = blockIdx.x;
  const int h   = blockIdx.y;
  const int tid = threadIdx.x;
  const int w   = tid >> 5;        // wave id 0..15
  const int l   = tid & 31;        // lane
  const int lh  = l >> 4;          // lane half: 0 = lanes 0-15, 1 = lanes 16-31
  const int lc  = l & 15;          // lane within half (WMMA column / M row)
  const int r0  = w * 16;          // this wave's row-tile base
  const float scale = 0.17677669529663687f;   // 1/sqrt(32)

  // ======== Phase 0a: async-DMA this head's weight slice into LDS (f32) ======
  // Rows 0..31 -> Wq[h], 32..63 -> Wk[h], 64..95 -> Wv[h]; 3072 b128 transfers.
#pragma unroll
  for (int it = 0; it < 6; ++it) {
    const int vec  = it * 512 + tid;            // float4 index 0..3071
    const int row  = vec >> 5;                  // 32 float4 per 128-f32 row
    const int colv = vec & 31;
    const int grow = (row >> 5) * CDIM + h * HD + (row & 31);
    async_g2l_b128(lds_byte_addr(uS.stage + row * CDIM + colv * 4),
                   qkvw + (size_t)grow * CDIM + colv * 4);
  }
  wait_asynccnt0();
  __syncthreads();

  // ======== Phase 0b: cooperative cvt f32 -> f16 weight cache ========
#pragma unroll
  for (int it = 0; it < 6; ++it) {
    const int vec = it * 512 + tid;
    const float4 f = ((const float4*)uS.stage)[vec];
    v4h hv; hv[0] = (_Float16)f.x; hv[1] = (_Float16)f.y;
            hv[2] = (_Float16)f.z; hv[3] = (_Float16)f.w;
    *(v4h*)(sW16 + vec * 4) = hv;
  }
  __syncthreads();

  _Float16* sQ = uS.qkv.q;
  _Float16* sK = uS.qkv.k;
  _Float16* sV = uS.qkv.v;

  // ======== Phase 1: QKV projection for this (b, h) via WMMA ========
  // out tiles j: 0,1 -> Q cols 0-15/16-31 ; 2,3 -> K ; 4,5 -> V
  v8f accp[6] = {};
  const int   arow = r0 + lc;                               // A row for this lane
  const float* xrow = x + ((size_t)b * NTOK + arow) * CDIM;
  __builtin_prefetch(xrow, 0, 1);                           // global_prefetch_b8

  const _Float16* wbase = sW16 + (size_t)lc * CDIM + (lh ? 16 : 0);
#pragma unroll
  for (int kc = 0; kc < 4; ++kc) {                          // K = 128 in 4x32
    const int ka = kc * 32 + (lh ? 16 : 0);
    const int kb = kc * 32 + (lh ? 24 : 8);
    v16h a  = afrag_f32(xrow + ka, xrow + kb);
    v16h bf = lds_bfrag(wbase + kc * 32);                   // j = 0 fragment
#pragma unroll
    for (int j = 0; j < 6; ++j) {
      v16h bn = bf;
      if (j < 5)                                            // prefetch j+1 fragment
        bn = lds_bfrag(wbase + (size_t)(((j + 1) >> 1) * 32 + ((j + 1) & 1) * 16) * CDIM
                             + kc * 32);
      accp[j] = wmma(a, bf, accp[j]);
      bf = bn;
    }
  }

  // Scatter C tiles into LDS (bias add; Q pre-scaled; V stored transposed).
#pragma unroll
  for (int j = 0; j < 6; ++j) {
    const int c = (j & 1) * 16 + lc;          // head-local column 0..31
#pragma unroll
    for (int i = 0; i < 8; ++i) {
      const int row = r0 + i + lh * 8;
      const float fv = accp[j][i];
      if (j < 2)      sQ[row * HD + c]   = (_Float16)((fv + qb[h * HD + c]) * scale);
      else if (j < 4) sK[row * HD + c]   = (_Float16)fv;
      else            sV[c * NTOK + row] = (_Float16)(fv + vb[h * HD + c]);
    }
  }
  __syncthreads();

  // ======== Phase 2: S = Q K^T (+rel-bias +mask), softmax ========
  v8f acc[16] = {};                           // 16x256 logits for this wave
  {
    const _Float16* q0 = sQ + (size_t)(r0 + lc) * HD + (lh ? 16 : 0);
    const _Float16* q1 = sQ + (size_t)(r0 + lc) * HD + (lh ? 24 : 8);
    v16h aq = lds_afrag(q0, q1);              // Q fragment reused for all 16 key tiles
    const _Float16* kbase = sK + (size_t)lc * HD + (lh ? 16 : 0);
    v16h bk = lds_bfrag(kbase);
#pragma unroll
    for (int t = 0; t < 16; ++t) {
      v16h bn = bk;
      if (t < 15) bn = lds_bfrag(kbase + (size_t)(t + 1) * 16 * HD);  // prefetch
      acc[t] = wmma(aq, bk, acc[t]);
      bk = bn;
    }
  }

  // rel-position bias (computed index, gathered from 25 KB table) + shift mask
  const float* mrow = mask + (size_t)(b & (NWIN - 1)) * NTOK * NTOK;
#pragma unroll
  for (int t = 0; t < 16; ++t) {
    const int col = t * 16 + lc;
    const int md = col >> 6, mh = (col >> 3) & 7, mw = col & 7;
#pragma unroll
    for (int i = 0; i < 8; ++i) {
      const int row = r0 + i + lh * 8;
      const int nd = row >> 6, nh = (row >> 3) & 7, nw = row & 7;
      const int idx = (nd - md + 3) * 225 + (nh - mh + 7) * 15 + (nw - mw + 7);
      acc[t][i] += table[idx * NH + h] + mrow[row * NTOK + col];
    }
  }

  // Row softmax: rows live in lane-halves; reduce across the 16 column-lanes.
  float rinv[8];
#pragma unroll
  for (int i = 0; i < 8; ++i) {
    float m = -3.4e38f;
#pragma unroll
    for (int t = 0; t < 16; ++t) m = fmaxf(m, acc[t][i]);
#pragma unroll
    for (int s = 1; s < 16; s <<= 1) m = fmaxf(m, __shfl_xor(m, s, 32));
    float sum = 0.f;
#pragma unroll
    for (int t = 0; t < 16; ++t) { float e = __expf(acc[t][i] - m); acc[t][i] = e; sum += e; }
#pragma unroll
    for (int s = 1; s < 16; s <<= 1) sum += __shfl_xor(sum, s, 32);
    rinv[i] = 1.f / sum;
  }

  // ======== Phase 3: O = P V via WMMA (P staged through per-wave LDS) ========
  v8f accO[2] = {};
  _Float16* myP = &sP[w][0];
#pragma unroll
  for (int ch = 0; ch < 8; ++ch) {            // 256 keys in 8 chunks of 32
    // V fragments are independent of P: hoist them ahead of the LDS bounce.
    v16h bv0 = lds_bfrag(sV + (size_t)(lc)      * NTOK + ch * 32 + (lh ? 16 : 0));
    v16h bv1 = lds_bfrag(sV + (size_t)(16 + lc) * NTOK + ch * 32 + (lh ? 16 : 0));
#pragma unroll
    for (int tt = 0; tt < 2; ++tt) {
      const int t = ch * 2 + tt;
#pragma unroll
      for (int i = 0; i < 8; ++i)
        myP[(i + lh * 8) * 32 + tt * 16 + lc] = (_Float16)(acc[t][i] * rinv[i]);
    }
    // Re-read the 16x32 P chunk as an A-fragment (same-wave DS ops are in-order).
    const _Float16* p0 = myP + lc * 32 + (lh ? 16 : 0);
    const _Float16* p1 = myP + lc * 32 + (lh ? 24 : 8);
    v16h ap = lds_afrag(p0, p1);
    accO[0] = wmma(ap, bv0, accO[0]);
    accO[1] = wmma(ap, bv1, accO[1]);
  }

  // ======== Store O: (B, N, C) with head offset ========
#pragma unroll
  for (int dt = 0; dt < 2; ++dt) {
#pragma unroll
    for (int i = 0; i < 8; ++i) {
      const int row = r0 + i + lh * 8;
      out[((size_t)b * NTOK + row) * CDIM + h * HD + dt * 16 + lc] = accO[dt][i];
    }
  }
}

extern "C" void kernel_launch(void* const* d_in, const int* in_sizes, int n_in,
                              void* d_out, int out_size, void* d_ws, size_t ws_size,
                              hipStream_t stream) {
  const float* x     = (const float*)d_in[0];
  const float* mask  = (const float*)d_in[1];
  const float* qkvw  = (const float*)d_in[2];
  const float* qb    = (const float*)d_in[3];
  const float* vb    = (const float*)d_in[4];
  const float* table = (const float*)d_in[5];
  float* out = (float*)d_out;
  (void)in_sizes; (void)n_in; (void)out_size; (void)d_ws; (void)ws_size;

  dim3 grid(512, NH);   // one workgroup per (window, head)
  dim3 block(512);      // 16 wave32's
  win_attn3d_kernel<<<grid, block, 0, stream>>>(x, mask, qkvw, qb, vb, table, out);
}
